// MelSpectrogramModel_4226247819477
// MI455X (gfx1250) — compile-verified, matrix operand order checked by reading, not run on hardware
//
#include <hip/hip_runtime.h>
#include <hip/hip_bf16.h>
#include <stdint.h>

#define WIN      2048
#define HOP      128
#define NFREQ    1025
#define NFPAD    1056            // 66 tiles of 16 -> 33 pairs
#define NMELS    128
#define NFRAMES  5152
#define TLEN     661376
#define BATCH    8
#define SPAN     18304           // samples staged per block: 127*128 + 2048
#define SPANB    (SPAN * 2)      // bytes per split part (bf16)

typedef __attribute__((ext_vector_type(8)))  float   v8f;
typedef __attribute__((ext_vector_type(8)))  __bf16  v8bf;
typedef __attribute__((ext_vector_type(16))) __bf16  v16bf;

static __device__ __forceinline__ v16bf cat8(v8bf a, v8bf b) {
  return __builtin_shufflevector(a, b, 0,1,2,3,4,5,6,7,8,9,10,11,12,13,14,15);
}
// XOR swizzle on 16B granules: byte-address stride of 256 between frames would
// alias all lanes onto the same LDS bank group; fold frame bits into [7:4].
static __device__ __forceinline__ unsigned swz(unsigned a) {
  return a ^ (((a >> 8) & 0xFu) << 4);
}

// ---------------- prep: f32 -> bf16 hi/lo splits ----------------

__global__ void split_signal(const float* __restrict__ x,
                             __bf16* __restrict__ xh, __bf16* __restrict__ xl, int n) {
  int i = blockIdx.x * blockDim.x + threadIdx.x;
  if (i >= n) return;
  float v = x[i];
  __bf16 h = (__bf16)v;
  xh[i] = h;
  xl[i] = (__bf16)(v - (float)h);
}

__global__ void split_weights(const float* __restrict__ cw, const float* __restrict__ sw,
                              __bf16* __restrict__ wch, __bf16* __restrict__ wcl,
                              __bf16* __restrict__ wsh, __bf16* __restrict__ wsl) {
  int i = blockIdx.x * blockDim.x + threadIdx.x;      // over NFPAD*WIN
  if (i >= NFPAD * WIN) return;
  int row = i / WIN, k = i - row * WIN;
  float c = 0.f, s = 0.f;
  if (row < NFREQ) { c = cw[row * WIN + k]; s = sw[row * WIN + k]; }
  __bf16 ch = (__bf16)c, sh = (__bf16)s;
  wch[i] = ch; wcl[i] = (__bf16)(c - (float)ch);
  wsh[i] = sh; wsl[i] = (__bf16)(s - (float)sh);
}

__global__ void split_mel(const float* __restrict__ mw,
                          __bf16* __restrict__ mh, __bf16* __restrict__ ml) {
  int i = blockIdx.x * blockDim.x + threadIdx.x;      // over NMELS*NFPAD
  if (i >= NMELS * NFPAD) return;
  int m = i / NFPAD, f = i - m * NFPAD;
  float v = (f < NFREQ) ? mw[m * NFREQ + f] : 0.f;
  __bf16 h = (__bf16)v;
  mh[i] = h; ml[i] = (__bf16)(v - (float)h);
}

// ---------------- fused frames*(cos,sin) -> power -> mel ----------------

__global__ __launch_bounds__(256) void mel_wmma(
    const __bf16* __restrict__ xh,  const __bf16* __restrict__ xl,
    const __bf16* __restrict__ wch, const __bf16* __restrict__ wcl,
    const __bf16* __restrict__ wsh, const __bf16* __restrict__ wsl,
    const __bf16* __restrict__ mh,  const __bf16* __restrict__ ml,
    float* __restrict__ out)
{
  __shared__ alignas(256) unsigned char smem[2 * SPANB];

  const int tid = threadIdx.x;
  const int b   = blockIdx.y;
  const int fb  = blockIdx.x;                       // 128-frame block
  const int base_local = fb * (HOP * 128);          // fb*16384 samples
  int span = SPAN;
  if (TLEN - base_local < span) span = TLEN - base_local;
  const size_t gbase = (size_t)b * TLEN + (size_t)base_local;

  // ---- async-stage signal window (hi & lo) into swizzled LDS ----
  const unsigned lds0 = (unsigned)(uintptr_t)(void*)&smem[0];
  {
    const __bf16* src0 = xh + gbase;
    const __bf16* src1 = xl + gbase;
    for (int i = tid * 8; i < span; i += 256 * 8) {
      unsigned off = (unsigned)(i * 2);
      unsigned la0 = lds0 + swz(off);
      unsigned la1 = lds0 + SPANB + swz(off);
      asm volatile("global_load_async_to_lds_b128 %0, %1, %2"
                   :: "v"(la0), "v"(off), "s"(src0) : "memory");
      asm volatile("global_load_async_to_lds_b128 %0, %1, %2"
                   :: "v"(la1), "v"(off), "s"(src1) : "memory");
    }
    asm volatile("s_wait_asynccnt 0" ::: "memory");
    __syncthreads();
  }

  const int wv   = tid >> 5;        // wave id: owns 16 frames
  const int lane = tid & 31;
  const int col  = lane & 15;       // frame column / matrix row within tile
  const int kh   = lane >> 4;       // K-half selector per WMMA layout

  int t0 = fb * 128 + wv * 16;
  if (t0 > NFRAMES - 16) t0 = NFRAMES - 16;         // tail block: duplicate work
  const int tloc = t0 - fb * 128;

  const unsigned fbase = (unsigned)((tloc + col) * 256);   // frame byte offset
  const unsigned char* sH = smem;
  const unsigned char* sL = smem + SPANB;

  v8f macc[8];
  const v8f vz = {0.f,0.f,0.f,0.f,0.f,0.f,0.f,0.f};
#pragma unroll
  for (int i = 0; i < 8; ++i) macc[i] = vz;

  for (int fp = 0; fp < NFPAD / 32; ++fp) {
    v16bf pbh, pbl;                 // packed power (hi/lo) B-operand, K=32 freqs
#pragma unroll
    for (int h = 0; h < 2; ++h) {
      const int ft = fp * 2 + h;
      v8f ac = vz, as = vz;
      const size_t arow = (size_t)(ft * 16 + col) * WIN + (size_t)(kh * 8);
      for (int kc = 0; kc < WIN / 32; ++kc) {
        const size_t ao = arow + (size_t)kc * 32;
        v16bf Ach = cat8(*(const v8bf*)(wch + ao), *(const v8bf*)(wch + ao + 16));
        v16bf Acl = cat8(*(const v8bf*)(wcl + ao), *(const v8bf*)(wcl + ao + 16));
        v16bf Ash = cat8(*(const v8bf*)(wsh + ao), *(const v8bf*)(wsh + ao + 16));
        v16bf Asl = cat8(*(const v8bf*)(wsl + ao), *(const v8bf*)(wsl + ao + 16));
        const unsigned kb = (unsigned)(kc * 64 + kh * 16);
        const unsigned o0 = swz(fbase + kb);
        const unsigned o1 = swz(fbase + kb + 32);
        v16bf Bh = cat8(*(const v8bf*)(sH + o0), *(const v8bf*)(sH + o1));
        v16bf Bl = cat8(*(const v8bf*)(sL + o0), *(const v8bf*)(sL + o1));
        // c += A*B with f32-via-bf16 split (drop lo*lo)
        ac = __builtin_amdgcn_wmma_f32_16x16x32_bf16(false, Ach, false, Bh, (short)0, ac, false, false);
        ac = __builtin_amdgcn_wmma_f32_16x16x32_bf16(false, Ach, false, Bl, (short)0, ac, false, false);
        ac = __builtin_amdgcn_wmma_f32_16x16x32_bf16(false, Acl, false, Bh, (short)0, ac, false, false);
        as = __builtin_amdgcn_wmma_f32_16x16x32_bf16(false, Ash, false, Bh, (short)0, as, false, false);
        as = __builtin_amdgcn_wmma_f32_16x16x32_bf16(false, Ash, false, Bl, (short)0, as, false, false);
        as = __builtin_amdgcn_wmma_f32_16x16x32_bf16(false, Asl, false, Bh, (short)0, as, false, false);
      }
      // power = c^2 + s^2 ; pack directly into mel B-operand layout
#pragma unroll
      for (int j = 0; j < 8; ++j) {
        float p = ac[j] * ac[j] + as[j] * as[j];
        __bf16 ph = (__bf16)p;
        pbh[h * 8 + j] = ph;
        pbl[h * 8 + j] = (__bf16)(p - (float)ph);
      }
    }
    // mel accumulation: mel[128 x 16 frames] += melW[:, 32 freqs] * power
    const size_t mcol = (size_t)fp * 32 + (size_t)(kh * 8);
#pragma unroll
    for (int mt = 0; mt < 8; ++mt) {
      const size_t mo = (size_t)(mt * 16 + col) * NFPAD + mcol;
      v16bf Mh = cat8(*(const v8bf*)(mh + mo), *(const v8bf*)(mh + mo + 16));
      v16bf Ml = cat8(*(const v8bf*)(ml + mo), *(const v8bf*)(ml + mo + 16));
      macc[mt] = __builtin_amdgcn_wmma_f32_16x16x32_bf16(false, Mh, false, pbh, (short)0, macc[mt], false, false);
      macc[mt] = __builtin_amdgcn_wmma_f32_16x16x32_bf16(false, Mh, false, pbl, (short)0, macc[mt], false, false);
      macc[mt] = __builtin_amdgcn_wmma_f32_16x16x32_bf16(false, Ml, false, pbh, (short)0, macc[mt], false, false);
    }
  }

  // write out[b][m][t]
  float* ob = out + (size_t)b * NMELS * NFRAMES;
#pragma unroll
  for (int mt = 0; mt < 8; ++mt) {
#pragma unroll
    for (int r = 0; r < 8; ++r) {
      const int m = mt * 16 + kh * 8 + r;
      ob[(size_t)m * NFRAMES + (size_t)(t0 + col)] = macc[mt][r];
    }
  }
}

// ---------------- host launcher ----------------

extern "C" void kernel_launch(void* const* d_in, const int* in_sizes, int n_in,
                              void* d_out, int out_size, void* d_ws, size_t ws_size,
                              hipStream_t stream) {
  (void)in_sizes; (void)n_in; (void)out_size; (void)ws_size;
  const float* x  = (const float*)d_in[0];
  const float* cw = (const float*)d_in[1];
  const float* sw = (const float*)d_in[2];
  const float* mw = (const float*)d_in[3];

  const size_t XN = (size_t)BATCH * TLEN;       // 5,291,008
  const size_t WN = (size_t)NFPAD * WIN;        // 2,162,688
  const size_t MN = (size_t)NMELS * NFPAD;      // 135,168
  __bf16* xh  = (__bf16*)d_ws;
  __bf16* xl  = xh + XN;
  __bf16* wch = xl + XN;
  __bf16* wcl = wch + WN;
  __bf16* wsh = wcl + WN;
  __bf16* wsl = wsh + WN;
  __bf16* mh  = wsl + WN;
  __bf16* ml  = mh + MN;                        // total ~39 MB of workspace

  split_signal <<<(unsigned)((XN + 255) / 256), 256, 0, stream>>>(x, xh, xl, (int)XN);
  split_weights<<<(unsigned)((WN + 255) / 256), 256, 0, stream>>>(cw, sw, wch, wcl, wsh, wsl);
  split_mel    <<<(unsigned)((MN + 255) / 256), 256, 0, stream>>>(mw, mh, ml);

  dim3 grid((NFRAMES + 127) / 128, BATCH);      // 41 x 8 blocks
  mel_wmma<<<grid, 256, 0, stream>>>(xh, xl, wch, wcl, wsh, wsl, mh, ml, (float*)d_out);
}
